// RoPEAttention_71339406786977
// MI455X (gfx1250) — matrix operand order repaired and checked by reading, not compiled, not run on hardware
//
#include <hip/hip_runtime.h>

// Problem constants (from reference)
#define Bn 2
#define Sn 2048
#define Hn 16
#define Dn 64
#define Cn 1024
#define F3n 3072
#define Mn (Bn * Sn)   // 4096 total rows

typedef __attribute__((ext_vector_type(16))) __bf16 v16bf;
typedef __attribute__((ext_vector_type(8)))  float  v8f;
typedef __attribute__((ext_vector_type(4)))  int    v4i;

union FragB16 { v16bf v; unsigned int p[8]; };

// CDNA5 async global->LDS copy (ASYNCcnt path), with portable fallback.
#if defined(__has_builtin)
#if __has_builtin(__builtin_amdgcn_global_load_async_to_lds_b128)
#define HAVE_ASYNC_LDS 1
#endif
#endif
#ifndef HAVE_ASYNC_LDS
#define HAVE_ASYNC_LDS 0
#endif

#if HAVE_ASYNC_LDS
typedef __attribute__((address_space(1))) v4i* gv4i_p;   // global
typedef __attribute__((address_space(3))) v4i* lv4i_p;   // LDS
#endif

__device__ __forceinline__ void async_copy16(const void* g, void* l) {
#if HAVE_ASYNC_LDS
  __builtin_amdgcn_global_load_async_to_lds_b128((gv4i_p)(void*)g, (lv4i_p)l, 0, 0);
#else
  *(uint4*)l = *(const uint4*)g;
#endif
}
__device__ __forceinline__ void async_join() {
#if HAVE_ASYNC_LDS
  asm volatile("s_wait_asynccnt 0" ::: "memory");
#endif
}

__device__ __forceinline__ unsigned short f2bf(float f) {
  unsigned int u = __float_as_uint(f);
  u += 0x7FFFu + ((u >> 16) & 1u);          // round-to-nearest-even
  return (unsigned short)(u >> 16);
}
__device__ __forceinline__ float bf2f(unsigned int lo16) {
  return __uint_as_float(lo16 << 16);
}

__device__ __forceinline__ v8f wmma_bf16(v16bf a, v16bf b, v8f c) {
  // (neg_a, A, neg_b, B, c_mod, C, reuse_a, reuse_b)
  return __builtin_amdgcn_wmma_f32_16x16x32_bf16(false, a, false, b, (short)0, c,
                                                 false, false);
}

// A-matrix fragment (16 rows x 32 k). Source storage: row-major, contiguous in k.
// ISA A layout: lane holds row = row0 + lane%16; VGPR j holds k-pair:
//   j<4: k = lh*8 + 2j ;  j>=4: k = 16 + lh*8 + 2(j-4)   (lh = lane/16)
__device__ __forceinline__ v16bf load_frag_a(const unsigned short* p, int stride,
                                             int row0, int kb) {
  const int lane = threadIdx.x & 31;
  const int lh = lane >> 4;
  const unsigned short* base = p + (long)(row0 + (lane & 15)) * stride + kb;
  FragB16 f;
#pragma unroll
  for (int j = 0; j < 8; ++j) {
    int k = ((j & 4) << 2) + (lh << 3) + ((j & 3) << 1);
    f.p[j] = *(const unsigned int*)(base + k);
  }
  return f.v;
}

// B-matrix fragment (32 k x 16 cols). Source storage T[col][k] (k contiguous).
// ISA B layout: lane holds col = col0 + lane%16; VGPR j holds k-pair:
//   j<4: k = lh*8 + 2j ;  j>=4: k = 16 + lh*8 + 2(j-4)   (lh = lane/16)
__device__ __forceinline__ v16bf load_frag_b(const unsigned short* p, int stride,
                                             int col0, int kb) {
  const int lane = threadIdx.x & 31;
  const int lh = lane >> 4;
  const unsigned short* base = p + (long)(col0 + (lane & 15)) * stride + kb;
  FragB16 f;
#pragma unroll
  for (int j = 0; j < 8; ++j) {
    int k = ((j & 4) << 2) + (lh << 3) + ((j & 3) << 1);
    f.p[j] = *(const unsigned int*)(base + k);
  }
  return f.v;
}

// ---------------- elementwise converts ----------------

__global__ void __launch_bounds__(256) xconv_kernel(const float* __restrict__ x,
                                                    unsigned short* __restrict__ xb,
                                                    int n4) {
  int i = blockIdx.x * 256 + threadIdx.x;
  if (i < n4) {
    float4 v = ((const float4*)x)[i];
    ushort4 o;
    o.x = f2bf(v.x); o.y = f2bf(v.y); o.z = f2bf(v.z); o.w = f2bf(v.w);
    ((ushort4*)xb)[i] = o;
  }
}

// Wt[n][k] = bf16(W[k][n]); block (32,8), 32x32 LDS tile transpose.
__global__ void __launch_bounds__(256) wtrans_kernel(const float* __restrict__ W,
                                                     unsigned short* __restrict__ Wt,
                                                     int K, int N) {
  __shared__ float t[32][33];
  const int n0 = blockIdx.x * 32, k0 = blockIdx.y * 32;
  const int tx = threadIdx.x, ty = threadIdx.y;
#pragma unroll
  for (int i = 0; i < 4; ++i)
    t[ty + i * 8][tx] = W[(long)(k0 + ty + i * 8) * N + n0 + tx];
  __syncthreads();
#pragma unroll
  for (int i = 0; i < 4; ++i)
    Wt[(long)(n0 + ty + i * 8) * K + k0 + tx] = f2bf(t[tx][ty + i * 8]);
}

// ---------------- tiled WMMA GEMM:  out = A(MxK) @ Wt^T + bias ----------------
// A bf16 row-major; Wt bf16 stored [n][k]; 128x128 block tile, BK=32, 8 waves,
// each wave computes a 32x64 sub-tile (2x4 WMMA fragments).

template <int OUT_F32>
__global__ void __launch_bounds__(256) gemm_bias_kernel(
    const unsigned short* __restrict__ A, const unsigned short* __restrict__ Wt,
    const float* __restrict__ bias, float* __restrict__ outf,
    unsigned short* __restrict__ outb, int M, int N, int K) {
  __shared__ unsigned short Alds[128 * 40];  // stride 40: 16B-aligned rows, bank spread
  __shared__ unsigned short Blds[128 * 40];
  const int tid = threadIdx.x;
  const int wave = tid >> 5, lane = tid & 31;
  const int wm = (wave >> 1) * 32;   // 4 waves along M
  const int wn = (wave & 1) * 64;    // 2 waves along N
  const int m0 = blockIdx.x * 128, n0 = blockIdx.y * 128;

  v8f z = {0.f, 0.f, 0.f, 0.f, 0.f, 0.f, 0.f, 0.f};
  v8f acc[2][4] = {{z, z, z, z}, {z, z, z, z}};

  for (int k0 = 0; k0 < K; k0 += 32) {
#pragma unroll
    for (int i = 0; i < 2; ++i) {               // A tile: 128x32 = 512 x 16B
      int idx = tid + i * 256;
      int row = idx >> 2, c8 = (idx & 3) * 8;
      async_copy16(A + (long)(m0 + row) * K + k0 + c8, &Alds[row * 40 + c8]);
    }
#pragma unroll
    for (int i = 0; i < 2; ++i) {               // B tile: 128x32 = 512 x 16B
      int idx = tid + i * 256;
      int row = idx >> 2, c8 = (idx & 3) * 8;
      async_copy16(Wt + (long)(n0 + row) * K + k0 + c8, &Blds[row * 40 + c8]);
    }
    async_join();
    __syncthreads();

    if (k0 + 32 < K) {                          // global_prefetch_b8 of next tiles
      __builtin_prefetch(A + (long)(m0 + (tid & 127)) * K + k0 + 32, 0, 1);
      __builtin_prefetch(Wt + (long)(n0 + (tid & 127)) * K + k0 + 32, 0, 1);
    }

    v16bf a0 = load_frag_a(Alds, 40, wm, 0);
    v16bf a1 = load_frag_a(Alds, 40, wm + 16, 0);
#pragma unroll
    for (int ni = 0; ni < 4; ++ni) {
      v16bf b = load_frag_b(Blds, 40, wn + ni * 16, 0);
      acc[0][ni] = wmma_bf16(a0, b, acc[0][ni]);
      acc[1][ni] = wmma_bf16(a1, b, acc[1][ni]);
    }
    __syncthreads();
  }

  const int lh = lane >> 4, ln = lane & 15;
#pragma unroll
  for (int mi = 0; mi < 2; ++mi)
#pragma unroll
    for (int ni = 0; ni < 4; ++ni) {
      int nn = n0 + wn + ni * 16 + ln;
      float bv = bias[nn];
#pragma unroll
      for (int r = 0; r < 8; ++r) {
        int mm = m0 + wm + mi * 16 + r + 8 * lh;
        float v = acc[mi][ni][r] + bv;
        if (OUT_F32) outf[(long)mm * N + nn] = v;
        else         outb[(long)mm * N + nn] = f2bf(v);
      }
    }
}

// ---------------- RoPE + split + reorder ----------------
// qkv bf16 (B,S,3,H,D) -> Q(B,H,S,D)*scale, K(B,H,S,D), Vt(B,H,D,S)

__global__ void __launch_bounds__(256) rope_split_kernel(
    const unsigned short* __restrict__ qkv, const float* __restrict__ freqs,
    unsigned short* __restrict__ Q, unsigned short* __restrict__ K,
    unsigned short* __restrict__ Vt) {
  int idx = blockIdx.x * 256 + threadIdx.x;     // (b, s, h, d2)
  int d2 = idx & 31;
  int h  = (idx >> 5) & 15;
  int s  = (idx >> 9) & 2047;
  int b  = idx >> 20;

  float fr = freqs[s * 32 + d2];
  float cs = cosf(fr), sn = sinf(fr);

  long base = ((long)(b * Sn + s)) * F3n + h * Dn + d2 * 2;
  unsigned int qp = *(const unsigned int*)(qkv + base);
  unsigned int kp = *(const unsigned int*)(qkv + base + Cn);
  unsigned int vp = *(const unsigned int*)(qkv + base + 2 * Cn);

  float qr = bf2f(qp & 0xffffu), qi = bf2f(qp >> 16);
  float kr = bf2f(kp & 0xffffu), ki = bf2f(kp >> 16);

  const float scale = 0.125f;   // 1/sqrt(64) folded into Q
  float oqr = (qr * cs - qi * sn) * scale;
  float oqi = (qr * sn + qi * cs) * scale;
  float okr = kr * cs - ki * sn;
  float oki = kr * sn + ki * cs;

  long qkbase = (((long)(b * Hn + h)) * Sn + s) * Dn + d2 * 2;
  *(unsigned int*)(Q + qkbase) =
      (unsigned int)f2bf(oqr) | ((unsigned int)f2bf(oqi) << 16);
  *(unsigned int*)(K + qkbase) =
      (unsigned int)f2bf(okr) | ((unsigned int)f2bf(oki) << 16);

  long vtbase = (((long)(b * Hn + h)) * Dn + d2 * 2) * Sn + s;
  Vt[vtbase]      = (unsigned short)(vp & 0xffffu);
  Vt[vtbase + Sn] = (unsigned short)(vp >> 16);
}

// ---------------- flash attention ----------------
// grid (S/128, B*H), 256 threads; each wave owns 16 query rows.

__global__ void __launch_bounds__(256) flash_attn_kernel(
    const unsigned short* __restrict__ Q, const unsigned short* __restrict__ Km,
    const unsigned short* __restrict__ Vt, unsigned short* __restrict__ O) {
  __shared__ unsigned short Klds[64 * 72];       // [key][d]
  __shared__ unsigned short Vlds[64 * 72];       // [d][key]
  __shared__ unsigned short Plds[8 * 16 * 72];   // per-wave P stage [row][key]
  const int tid = threadIdx.x;
  const int wave = tid >> 5, lane = tid & 31;
  const int lh = lane >> 4, ln = lane & 15;
  const int bh = blockIdx.y;
  const int q0 = blockIdx.x * 128 + wave * 16;

  const unsigned short* Qh = Q  + (long)bh * Sn * Dn;
  const unsigned short* Kh = Km + (long)bh * Sn * Dn;
  const unsigned short* Vh = Vt + (long)bh * Dn * Sn;

  v16bf aq0 = load_frag_a(Qh, Dn, q0, 0);    // Q rows once, d=0..31
  v16bf aq1 = load_frag_a(Qh, Dn, q0, 32);   // d=32..63

  v8f z = {0.f, 0.f, 0.f, 0.f, 0.f, 0.f, 0.f, 0.f};
  v8f acc[4] = {z, z, z, z};
  float m[8], l[8];
#pragma unroll
  for (int r = 0; r < 8; ++r) { m[r] = -1e30f; l[r] = 0.f; }

  unsigned short* Pw = &Plds[wave * 16 * 72];

  for (int kv0 = 0; kv0 < Sn; kv0 += 64) {
#pragma unroll
    for (int i = 0; i < 2; ++i) {            // stage K and Vt 64x64 tiles
      int idx = tid + i * 256;
      int row = idx >> 2, c8 = (idx & 3) * 8;
      async_copy16(Kh + (long)(kv0 + row) * Dn + c8, &Klds[row * 72 + c8]);
      async_copy16(Vh + (long)row * Sn + kv0 + c8,   &Vlds[row * 72 + c8]);
    }
    async_join();
    __syncthreads();

    // S = Q K^T  (scale pre-folded into Q)
    v8f sc[4];
#pragma unroll
    for (int f = 0; f < 4; ++f) {
      v16bf bk0 = load_frag_b(Klds, 72, f * 16, 0);
      v16bf bk1 = load_frag_b(Klds, 72, f * 16, 32);
      sc[f] = wmma_bf16(aq0, bk0, z);
      sc[f] = wmma_bf16(aq1, bk1, sc[f]);
    }

    // online softmax (row r lives in VGPR r across the 16-lane half-group)
#pragma unroll
    for (int r = 0; r < 8; ++r) {
      float v = fmaxf(fmaxf(sc[0][r], sc[1][r]), fmaxf(sc[2][r], sc[3][r]));
      v = fmaxf(v, __shfl_xor(v, 1, 32));
      v = fmaxf(v, __shfl_xor(v, 2, 32));
      v = fmaxf(v, __shfl_xor(v, 4, 32));
      v = fmaxf(v, __shfl_xor(v, 8, 32));
      float nm = fmaxf(m[r], v);
      float alpha = __expf(m[r] - nm);
      m[r] = nm;
      l[r] *= alpha;
#pragma unroll
      for (int df = 0; df < 4; ++df) acc[df][r] *= alpha;
#pragma unroll
      for (int f = 0; f < 4; ++f) sc[f][r] = __expf(sc[f][r] - nm);
      float rs = sc[0][r] + sc[1][r] + sc[2][r] + sc[3][r];
      rs += __shfl_xor(rs, 1, 32);
      rs += __shfl_xor(rs, 2, 32);
      rs += __shfl_xor(rs, 4, 32);
      rs += __shfl_xor(rs, 8, 32);
      l[r] += rs;
    }

    // stage P (C-layout -> A-layout transpose through wave-private LDS)
#pragma unroll
    for (int f = 0; f < 4; ++f)
#pragma unroll
      for (int r = 0; r < 8; ++r)
        Pw[(r + 8 * lh) * 72 + f * 16 + ln] = f2bf(sc[f][r]);
    asm volatile("s_wait_dscnt 0" ::: "memory");   // wave-local LDS RAW

    v16bf ap0 = load_frag_a(Pw, 72, 0, 0);
    v16bf ap1 = load_frag_a(Pw, 72, 0, 32);
#pragma unroll
    for (int df = 0; df < 4; ++df) {               // O += P V
      v16bf bv0 = load_frag_b(Vlds, 72, df * 16, 0);
      v16bf bv1 = load_frag_b(Vlds, 72, df * 16, 32);
      acc[df] = wmma_bf16(ap0, bv0, acc[df]);
      acc[df] = wmma_bf16(ap1, bv1, acc[df]);
    }
    __syncthreads();
  }

  const int b = bh >> 4, h = bh & 15;
#pragma unroll
  for (int df = 0; df < 4; ++df) {
    int d = df * 16 + ln;
#pragma unroll
    for (int r = 0; r < 8; ++r) {
      int srow = q0 + r + 8 * lh;
      float o = acc[df][r] / l[r];
      O[(((long)(b * Sn + srow)) * Hn + h) * Dn + d] = f2bf(o);
    }
  }
}

// ---------------- host launch ----------------

extern "C" void kernel_launch(void* const* d_in, const int* in_sizes, int n_in,
                              void* d_out, int out_size, void* d_ws, size_t ws_size,
                              hipStream_t stream) {
  (void)in_sizes; (void)n_in; (void)out_size; (void)ws_size;
  const float* x     = (const float*)d_in[0];
  const float* freqs = (const float*)d_in[1];
  const float* W_qkv = (const float*)d_in[2];
  const float* b_qkv = (const float*)d_in[3];
  const float* W_out = (const float*)d_in[4];
  const float* b_out = (const float*)d_in[5];
  float* out = (float*)d_out;

  char* ws = (char*)d_ws;
  size_t off = 0;
  auto alloc = [&](size_t bytes) {
    char* p = ws + off;
    off += (bytes + 255) & ~(size_t)255;
    return p;
  };
  unsigned short* Xbf   = (unsigned short*)alloc((size_t)Mn * Cn * 2);    // 8 MB
  unsigned short* Wtqkv = (unsigned short*)alloc((size_t)F3n * Cn * 2);   // 6 MB
  unsigned short* Wtout = (unsigned short*)alloc((size_t)Cn * Cn * 2);    // 2 MB
  unsigned short* QKVb  = (unsigned short*)alloc((size_t)Mn * F3n * 2);   // 24 MB
  unsigned short* Qb    = (unsigned short*)alloc((size_t)Mn * Cn * 2);    // 8 MB
  unsigned short* Kb    = (unsigned short*)alloc((size_t)Mn * Cn * 2);    // 8 MB
  unsigned short* Vtb   = (unsigned short*)alloc((size_t)Mn * Cn * 2);    // 8 MB
  unsigned short* Ob    = (unsigned short*)alloc((size_t)Mn * Cn * 2);    // 8 MB

  xconv_kernel<<<(Mn * Cn / 4 + 255) / 256, 256, 0, stream>>>(x, Xbf, Mn * Cn / 4);
  wtrans_kernel<<<dim3(F3n / 32, Cn / 32), dim3(32, 8), 0, stream>>>(W_qkv, Wtqkv, Cn, F3n);
  wtrans_kernel<<<dim3(Cn / 32, Cn / 32), dim3(32, 8), 0, stream>>>(W_out, Wtout, Cn, Cn);

  gemm_bias_kernel<0><<<dim3(Mn / 128, F3n / 128), 256, 0, stream>>>(
      Xbf, Wtqkv, b_qkv, nullptr, QKVb, Mn, F3n, Cn);

  rope_split_kernel<<<(Bn * Sn * Hn * (Dn / 2)) / 256, 256, 0, stream>>>(
      QKVb, freqs, Qb, Kb, Vtb);

  flash_attn_kernel<<<dim3(Sn / 128, Bn * Hn), 256, 0, stream>>>(Qb, Kb, Vtb, Ob);

  gemm_bias_kernel<1><<<dim3(Mn / 128, Cn / 128), 256, 0, stream>>>(
      Ob, Wtout, b_out, out, nullptr, Mn, Cn, Cn);
}
